// HaloAttention_3289944949142
// MI455X (gfx1250) — matrix operand order repaired and checked by reading, not compiled, not run on hardware
//
#include <hip/hip_runtime.h>
#include <hip/hip_bf16.h>
#include <math.h>

typedef __attribute__((ext_vector_type(16))) __bf16 v16bf;
typedef __attribute__((ext_vector_type(8)))  __bf16 v8bf;
typedef __attribute__((ext_vector_type(4)))  __bf16 v4bf;
typedef __attribute__((ext_vector_type(2)))  __bf16 v2bf;
typedef __attribute__((ext_vector_type(8)))  float  v8f;

#define WMMA_BF16(a, b, c) \
  __builtin_amdgcn_wmma_f32_16x16x32_bf16(false, (a), false, (b), (short)0, (c), false, false)

#define CDIM   512
#define NPIX   32768                // 8 * 64 * 64
#define NEGMAX (-3.402823466e38f)
#define LDA    40                   // padded LDS strides (halves): 80B rows, 16B aligned
#define LDB    40
#define LDV    264

__device__ __forceinline__ v8f v8f_zero() {
  v8f z = {0.f, 0.f, 0.f, 0.f, 0.f, 0.f, 0.f, 0.f};
  return z;
}
__device__ __forceinline__ v8bf v8bf_zero() {
  v8bf z;
#pragma unroll
  for (int e = 0; e < 8; ++e) z[e] = (__bf16)0.f;
  return z;
}

// ---- WMMA fragment loaders (wave32, 16x16x32 bf16) ----
// A (16x32 MxK) from row-major tile: lane 0-15 row=lane, K={0..7,16..23};
// lane 16-31 row=lane-16, K={8..15,24..31}. Two contiguous 16B runs -> 2x ds_load_b128.
__device__ __forceinline__ v16bf load_a(const __bf16* s, int row0, int k0, int ld, int lane) {
  int r  = row0 + (lane & 15);
  int kb = k0 + ((lane >> 4) << 3);
  const __bf16* p = s + r * ld + kb;
  v16bf a;
#pragma unroll
  for (int e = 0; e < 16; ++e) a[e] = p[(e & 7) + ((e >> 3) << 4)];
  return a;
}

// B (32x16 KxN) where B[k][n] = T[n][k], T row-major NxK (stride ld):
// lane holds column n, K={0..7,16..23} (lanes 0-15) / {8..15,24..31} (lanes 16-31) -> 2x ds_load_b128.
__device__ __forceinline__ v16bf load_b_nk(const __bf16* s, int k0, int n0, int ld, int lane) {
  int n  = n0 + (lane & 15);
  int kb = k0 + ((lane >> 4) << 3);
  const __bf16* p = s + n * ld + kb;
  v16bf b;
#pragma unroll
  for (int e = 0; e < 16; ++e) b[e] = p[(e & 7) + ((e >> 3) << 4)];
  return b;
}

// ------------------------------------------------------------------
// Kernel 1: qk[p, 0:512] = x_pix @ (W_q * 0.125); qk[p, 512:1024] = x_pix @ W_k
//           Vt[b][ch][pix] = (x_pix @ W_v) transposed to channel-major
// M = 32768, K = 512, N = 1536. Tile 128x128, 256 threads (8 waves).
// ------------------------------------------------------------------
__global__ __launch_bounds__(256) void qkv_proj_kernel(
    const float* __restrict__ x, const float* __restrict__ Wq,
    const float* __restrict__ Wkv, __bf16* __restrict__ qk,
    __bf16* __restrict__ Vt) {
  __shared__ __bf16 As[128 * LDA];    // 10 KB
  __shared__ __bf16 BsT[128 * LDB];   // 10 KB (transposed weight tile)
  __shared__ __bf16 Cs[128 * 128];    // 32 KB

  const int tid  = threadIdx.x;
  const int wave = tid >> 5, lane = tid & 31;
  const int m0  = blockIdx.x * 128;
  const int n0g = blockIdx.y * 128;
  const int wm = (wave & 3) * 32;
  const int wn = (wave >> 2) * 64;
  const bool isV = (n0g >= 1024);

  v8f acc[2][4];
#pragma unroll
  for (int i = 0; i < 2; ++i)
#pragma unroll
    for (int j = 0; j < 4; ++j) acc[i][j] = v8f_zero();

  for (int kc = 0; kc < CDIM; kc += 32) {
    __syncthreads();
    // ---- A staging: gather 8 k-pairs, then packed b32 stores ----
    float a0[8], a1[8];
#pragma unroll
    for (int it = 0; it < 8; ++it) {
      int j = it * 256 + tid;
      int m = j & 127, kp = j >> 7;         // kp 0..15
      int p = m0 + m;
      int bimg = p >> 12, sp = p & 4095;
      size_t base = (size_t)bimg * (CDIM * 4096) + sp;
      a0[it] = x[base + (size_t)(kc + 2 * kp) * 4096];
      a1[it] = x[base + (size_t)(kc + 2 * kp + 1) * 4096];
    }
    if (kc + 32 < CDIM) {  // prefetch next A chunk (global_prefetch_b8)
      int p = m0 + (tid & 127);
      int bimg = p >> 12, sp = p & 4095;
      __builtin_prefetch(&x[(size_t)bimg * (CDIM * 4096) +
                            (size_t)(kc + 32 + (tid >> 7)) * 4096 + sp], 0, 1);
    }
#pragma unroll
    for (int it = 0; it < 8; ++it) {
      int j = it * 256 + tid;
      int m = j & 127, kp = j >> 7;
      v2bf t;
      t[0] = (__bf16)a0[it];
      t[1] = (__bf16)a1[it];
      *(v2bf*)&As[m * LDA + 2 * kp] = t;
    }
    // ---- B staging (transposed): 4 groups of 4 consecutive k ----
    float w[16];
#pragma unroll
    for (int g = 0; g < 4; ++g) {
      int j = g * 256 + tid;
      int n = j & 127, kg = j >> 7;         // kg 0..7
      int gn = n0g + n;
#pragma unroll
      for (int u = 0; u < 4; ++u) {
        int k = kc + kg * 4 + u;
        w[g * 4 + u] = (gn < 512) ? Wq[k * 512 + gn] * 0.125f
                                  : Wkv[k * 1024 + (gn - 512)];
      }
    }
#pragma unroll
    for (int g = 0; g < 4; ++g) {
      int j = g * 256 + tid;
      int n = j & 127, kg = j >> 7;
      v4bf t;
#pragma unroll
      for (int u = 0; u < 4; ++u) t[u] = (__bf16)w[g * 4 + u];
      *(v4bf*)&BsT[n * LDB + kg * 4] = t;
    }
    __syncthreads();

    v16bf af[2], bfr[4];
#pragma unroll
    for (int i = 0; i < 2; ++i) af[i] = load_a(As, wm + i * 16, 0, LDA, lane);
#pragma unroll
    for (int j = 0; j < 4; ++j) bfr[j] = load_b_nk(BsT, 0, wn + j * 16, LDB, lane);
#pragma unroll
    for (int i = 0; i < 2; ++i)
#pragma unroll
      for (int j = 0; j < 4; ++j) acc[i][j] = WMMA_BF16(af[i], bfr[j], acc[i][j]);
  }

  __syncthreads();
#pragma unroll
  for (int i = 0; i < 2; ++i)
#pragma unroll
    for (int j = 0; j < 4; ++j) {
      int r0 = wm + i * 16 + ((lane >> 4) << 3);
      int c0 = wn + j * 16 + (lane & 15);
#pragma unroll
      for (int v = 0; v < 8; ++v) Cs[(r0 + v) * 128 + c0] = (__bf16)acc[i][j][v];
    }
  __syncthreads();

  if (!isV) {
    // row-major qk writeout, 16B chunks, coalesced in n
#pragma unroll
    for (int it = 0; it < 8; ++it) {
      int ch = it * 256 + tid;               // 2048 chunks of 8
      int m = ch >> 4, c8 = (ch & 15) * 8;
      *(v8bf*)&qk[(size_t)(m0 + m) * 1024 + n0g + c8] = *(const v8bf*)&Cs[m * 128 + c8];
    }
  } else {
    // channel-major Vt writeout, coalesced in pixels
#pragma unroll
    for (int it = 0; it < 8; ++it) {
      int ch = it * 256 + tid;
      int mg = ch & 15, n = ch >> 4;         // n 0..127
      int m = mg * 8;
      v8bf t;
#pragma unroll
      for (int u = 0; u < 8; ++u) t[u] = Cs[(m + u) * 128 + n];
      int p = m0 + m;
      int bimg = p >> 12, sp = p & 4095;
      *(v8bf*)&Vt[(size_t)bimg * (512 * 4096) + (size_t)(n0g - 1024 + n) * 4096 + sp] = t;
    }
  }
}

// ------------------------------------------------------------------
// Kernel 2: per (window, head) fused halo attention.
// grid = 512 windows * 8 heads = 4096 blocks, 256 threads (8 waves).
// ------------------------------------------------------------------
__global__ __launch_bounds__(256) void halo_attn_kernel(
    const __bf16* __restrict__ qk, const __bf16* __restrict__ Vt,
    const float* __restrict__ relh, const float* __restrict__ relw,
    __bf16* __restrict__ O) {
  __shared__ __bf16 Qs[64 * 72];      // 9 KB  (reused: softmax partials, O tile)
  __shared__ __bf16 Ks[256 * 72];     // 36 KB (reused: attn bf16 64x256)
  __shared__ __bf16 VsT[64 * LDV];    // 33 KB (V transposed: [d][kpix])
  __shared__ __bf16 RWb[32 * 64];     // 4 KB
  __shared__ __bf16 RHb[32 * 64];     // 4 KB
  __shared__ float  RWs[64 * 32];     // 8 KB
  __shared__ float  RHs[64 * 32];     // 8 KB
  __shared__ float  sim[64 * 256];    // 64 KB

  const int tid  = threadIdx.x;
  const int wave = tid >> 5, lane = tid & 31;
  const int wh   = blockIdx.x;
  const int head = wh & 7;
  const int win  = wh >> 3;
  const int wx = win & 7, wy = (win >> 3) & 7, bimg = win >> 6;
  const int cq = head * 64;
  const size_t rowbase = (size_t)bimg * 4096;

  // ---- stage Q (64x64): vectorized copy ----
  {
    v8bf t[2];
#pragma unroll
    for (int it = 0; it < 2; ++it) {
      int ch = it * 256 + tid;
      int r = ch >> 3, c8 = (ch & 7) * 8;
      int p = (int)rowbase + (wy * 8 + (r >> 3)) * 64 + (wx * 8 + (r & 7));
      t[it] = *(const v8bf*)&qk[(size_t)p * 1024 + cq + c8];
    }
#pragma unroll
    for (int it = 0; it < 2; ++it) {
      int ch = it * 256 + tid;
      int r = ch >> 3, c8 = (ch & 7) * 8;
      *(v8bf*)&Qs[r * 72 + c8] = t[it];
    }
  }
  // ---- stage K (256x64): predicated vector copy, zero halo ----
  {
    v8bf t[8];
#pragma unroll
    for (int it = 0; it < 8; ++it) {
      int ch = it * 256 + tid;
      int r = ch >> 3, c8 = (ch & 7) * 8;
      int gy = wy * 8 + (r >> 4) - 4, gx = wx * 8 + (r & 15) - 4;
      if ((unsigned)gy < 64u && (unsigned)gx < 64u)
        t[it] = *(const v8bf*)&qk[(rowbase + (size_t)(gy * 64 + gx)) * 1024 + 512 + cq + c8];
      else
        t[it] = v8bf_zero();
    }
#pragma unroll
    for (int it = 0; it < 8; ++it) {
      int ch = it * 256 + tid;
      int r = ch >> 3, c8 = (ch & 7) * 8;
      *(v8bf*)&Ks[r * 72 + c8] = t[it];
    }
  }
  // ---- stage V transposed: VsT[d][kpix] from channel-major Vt ----
  {
#pragma unroll
    for (int it = 0; it < 4; ++it) {
      int ch = it * 256 + tid;               // (d, ky) pairs: 64*16
      int d = ch >> 4, ky = ch & 15;
      int gy = wy * 8 + ky - 4;
      v8bf t0 = v8bf_zero(), t1 = v8bf_zero();
      if ((unsigned)gy < 64u) {
        const __bf16* src = Vt + (size_t)bimg * (512 * 4096) + (size_t)(cq + d) * 4096 + gy * 64;
#pragma unroll
        for (int kx = 0; kx < 8; ++kx) {
          int gx = wx * 8 + kx - 4;
          t0[kx] = ((unsigned)gx < 64u) ? src[gx] : (__bf16)0.f;
        }
#pragma unroll
        for (int kx = 8; kx < 16; ++kx) {
          int gx = wx * 8 + kx - 4;
          t1[kx - 8] = ((unsigned)gx < 64u) ? src[gx] : (__bf16)0.f;
        }
      }
      *(v8bf*)&VsT[d * LDV + ky * 16] = t0;
      *(v8bf*)&VsT[d * LDV + ky * 16 + 8] = t1;
    }
  }
  // ---- rel tables (31 rows + zero pad), pair-packed ----
  {
#pragma unroll
    for (int it = 0; it < 4; ++it) {
      int j = it * 256 + tid;                // 1024 pairs per table
      int rr = j >> 5, dp = (j & 31) * 2;
      float w0 = 0.f, w1 = 0.f, h0 = 0.f, h1 = 0.f;
      if (rr < 31) {
        w0 = relw[rr * 64 + dp]; w1 = relw[rr * 64 + dp + 1];
        h0 = relh[rr * 64 + dp]; h1 = relh[rr * 64 + dp + 1];
      }
      v2bf tw; tw[0] = (__bf16)w0; tw[1] = (__bf16)w1;
      v2bf th; th[0] = (__bf16)h0; th[1] = (__bf16)h1;
      *(v2bf*)&RWb[rr * 64 + dp] = tw;
      *(v2bf*)&RHb[rr * 64 + dp] = th;
    }
  }
  __syncthreads();

  // ---- rel-logit tables: RW/RH[r][j] = q_r . rel[j] (waves 0-3 W, 4-7 H) ----
  {
    const __bf16* RB = (wave < 4) ? RWb : RHb;
    float* RS = (wave < 4) ? RWs : RHs;
    int r0 = (wave & 3) * 16;
    v8f racc[2] = {v8f_zero(), v8f_zero()};
#pragma unroll
    for (int kc = 0; kc < 64; kc += 32) {
      v16bf a = load_a(Qs, r0, kc, 72, lane);
#pragma unroll
      for (int j = 0; j < 2; ++j) {
        v16bf b = load_b_nk(RB, kc, j * 16, 64, lane);
        racc[j] = WMMA_BF16(a, b, racc[j]);
      }
    }
    int rr = r0 + ((lane >> 4) << 3), cc = lane & 15;
#pragma unroll
    for (int j = 0; j < 2; ++j)
#pragma unroll
      for (int v = 0; v < 8; ++v) RS[(rr + v) * 32 + (j * 16 + cc)] = racc[j][v];
  }

  // ---- sim = Q @ K^T : wave computes 64 rows x 32 cols ----
  const int n0 = wave * 32;
  v8f sacc[4][2];
#pragma unroll
  for (int i = 0; i < 4; ++i) { sacc[i][0] = v8f_zero(); sacc[i][1] = v8f_zero(); }
#pragma unroll
  for (int kc = 0; kc < 64; kc += 32) {
    v16bf b0 = load_b_nk(Ks, kc, n0, 72, lane);
    v16bf b1 = load_b_nk(Ks, kc, n0 + 16, 72, lane);
#pragma unroll
    for (int i = 0; i < 4; ++i) {
      v16bf a = load_a(Qs, i * 16, kc, 72, lane);
      sacc[i][0] = WMMA_BF16(a, b0, sacc[i][0]);
      sacc[i][1] = WMMA_BF16(a, b1, sacc[i][1]);
    }
  }
  __syncthreads();   // RWs/RHs complete; Ks free after this

  // ---- assemble sim + rel logits + (inverted) validity mask ----
  {
    int cc = lane & 15, rb = (lane >> 4) << 3;
#pragma unroll
    for (int i = 0; i < 4; ++i)
#pragma unroll
      for (int j = 0; j < 2; ++j) {
        int kcol = n0 + j * 16 + cc;
        int ky = kcol >> 4, kx = kcol & 15;
        int gy = wy * 8 + ky - 4, gx = wx * 8 + kx - 4;
        bool valid = ((unsigned)gy < 64u) && ((unsigned)gx < 64u);
#pragma unroll
        for (int v = 0; v < 8; ++v) {
          int r = i * 16 + rb + v;
          float sv = sacc[i][j][v]
                   + RWs[r * 32 + (kx - (r & 7) + 15)]
                   + RHs[r * 32 + (ky - (r >> 3) + 15)];
          sim[r * 256 + kcol] = valid ? NEGMAX : sv;   // reference's inverted mask
        }
      }
  }
  __syncthreads();

  // ---- softmax over 256 (4 threads/row), vectorized; attn bf16 -> Ks ----
  {
    float* part = (float*)Qs;
    __bf16* attn = Ks;
    int r = tid >> 2, q4 = tid & 3;
    float* srow = &sim[r * 256 + q4 * 64];
    float4* s4 = (float4*)srow;
    float mx = NEGMAX;
#pragma unroll
    for (int i = 0; i < 16; ++i) {
      float4 t = s4[i];
      mx = fmaxf(mx, fmaxf(fmaxf(t.x, t.y), fmaxf(t.z, t.w)));
    }
    part[tid] = mx;
    __syncthreads();
    float rmx = fmaxf(fmaxf(part[r * 4 + 0], part[r * 4 + 1]),
                      fmaxf(part[r * 4 + 2], part[r * 4 + 3]));
    float s = 0.f;
#pragma unroll
    for (int i = 0; i < 16; ++i) {
      float4 t = s4[i];
      t.x = __expf(t.x - rmx); t.y = __expf(t.y - rmx);
      t.z = __expf(t.z - rmx); t.w = __expf(t.w - rmx);
      s += t.x + t.y + t.z + t.w;
      s4[i] = t;
    }
    __syncthreads();
    part[tid] = s;
    __syncthreads();
    float rs = part[r * 4 + 0] + part[r * 4 + 1] + part[r * 4 + 2] + part[r * 4 + 3];
    float inv = 1.f / rs;
#pragma unroll
    for (int i = 0; i < 16; ++i) {
      float4 t = s4[i];
      v4bf o;
      o[0] = (__bf16)(t.x * inv); o[1] = (__bf16)(t.y * inv);
      o[2] = (__bf16)(t.z * inv); o[3] = (__bf16)(t.w * inv);
      *(v4bf*)&attn[r * 256 + q4 * 64 + i * 4] = o;
    }
  }
  __syncthreads();

  // ---- out = attn(64x256) @ V(256x64): wave -> 16 rows x 32 cols ----
  {
    const __bf16* attn = Ks;
    __bf16* Ob = Qs;                        // reuse Qs as 64x64 output tile
    int mt = wave & 3;
    int nb = (wave >> 2) * 32;
    v8f oacc[2] = {v8f_zero(), v8f_zero()};
#pragma unroll
    for (int kc = 0; kc < 256; kc += 32) {
      v16bf a = load_a(attn, mt * 16, kc, 256, lane);
#pragma unroll
      for (int j = 0; j < 2; ++j) {
        v16bf b = load_b_nk(VsT, kc, nb + j * 16, LDV, lane);
        oacc[j] = WMMA_BF16(a, b, oacc[j]);
      }
    }
    int rr = mt * 16 + ((lane >> 4) << 3), cc = lane & 15;
#pragma unroll
    for (int j = 0; j < 2; ++j)
#pragma unroll
      for (int v = 0; v < 8; ++v)
        Ob[(rr + v) * 64 + nb + j * 16 + cc] = (__bf16)oacc[j][v];
  }
  __syncthreads();
  // coalesced O writeout
  {
    const __bf16* Ob = Qs;
#pragma unroll
    for (int it = 0; it < 2; ++it) {
      int ch = it * 256 + tid;               // 512 chunks of 8
      int r = ch >> 3, c8 = (ch & 7) * 8;
      size_t p = rowbase + (size_t)((wy * 8 + (r >> 3)) * 64 + (wx * 8 + (r & 7)));
      *(v8bf*)&O[p * 512 + head * 64 + c8] = *(const v8bf*)&Ob[r * 64 + c8];
    }
  }
}

// ------------------------------------------------------------------
// Kernel 3: out = O(32768x512)bf16 @ W_out + b_out, scatter to (b,c,h,w) f32
// ------------------------------------------------------------------
__global__ __launch_bounds__(256) void out_proj_kernel(
    const __bf16* __restrict__ O, const float* __restrict__ Wout,
    const float* __restrict__ bout, float* __restrict__ out) {
  __shared__ __bf16 As[128 * LDA];
  __shared__ __bf16 BsT[128 * LDB];
  __shared__ float  Cs[128 * 128];    // 64 KB

  const int tid  = threadIdx.x;
  const int wave = tid >> 5, lane = tid & 31;
  const int m0  = blockIdx.x * 128;
  const int n0g = blockIdx.y * 128;
  const int wm = (wave & 3) * 32;
  const int wn = (wave >> 2) * 64;

  v8f acc[2][4];
#pragma unroll
  for (int i = 0; i < 2; ++i)
#pragma unroll
    for (int j = 0; j < 4; ++j) acc[i][j] = v8f_zero();

  for (int kc = 0; kc < 512; kc += 32) {
    __syncthreads();
    // A staging: O is already bf16, 16B vector copies
    {
      v8bf t[2];
#pragma unroll
      for (int it = 0; it < 2; ++it) {
        int ch = it * 256 + tid;             // 512 chunks of 8
        int m = ch >> 2, k8 = (ch & 3) * 8;
        t[it] = *(const v8bf*)&O[(size_t)(m0 + m) * 512 + kc + k8];
      }
#pragma unroll
      for (int it = 0; it < 2; ++it) {
        int ch = it * 256 + tid;
        int m = ch >> 2, k8 = (ch & 3) * 8;
        *(v8bf*)&As[m * LDA + k8] = t[it];
      }
    }
    // B staging (transposed)
    {
      float w[16];
#pragma unroll
      for (int g = 0; g < 4; ++g) {
        int j = g * 256 + tid;
        int n = j & 127, kg = j >> 7;
#pragma unroll
        for (int u = 0; u < 4; ++u)
          w[g * 4 + u] = Wout[(kc + kg * 4 + u) * 512 + n0g + n];
      }
#pragma unroll
      for (int g = 0; g < 4; ++g) {
        int j = g * 256 + tid;
        int n = j & 127, kg = j >> 7;
        v4bf t;
#pragma unroll
        for (int u = 0; u < 4; ++u) t[u] = (__bf16)w[g * 4 + u];
        *(v4bf*)&BsT[n * LDB + kg * 4] = t;
      }
    }
    __syncthreads();

    v16bf af[2], bfr[4];
#pragma unroll
    for (int i = 0; i < 2; ++i) af[i] = load_a(As, wm + i * 16, 0, LDA, lane);
#pragma unroll
    for (int j = 0; j < 4; ++j) bfr[j] = load_b_nk(BsT, 0, wn + j * 16, LDB, lane);
#pragma unroll
    for (int i = 0; i < 2; ++i)
#pragma unroll
      for (int j = 0; j < 4; ++j) acc[i][j] = WMMA_BF16(af[i], bfr[j], acc[i][j]);
  }

  __syncthreads();
#pragma unroll
  for (int i = 0; i < 2; ++i)
#pragma unroll
    for (int j = 0; j < 4; ++j) {
      int r0 = wm + i * 16 + ((lane >> 4) << 3);
      int c0 = wn + j * 16 + (lane & 15);
#pragma unroll
      for (int v = 0; v < 8; ++v) Cs[(r0 + v) * 128 + c0] = acc[i][j][v];
    }
  __syncthreads();
  // channel-major f32 scatter, float4 chunks coalesced over pixels
#pragma unroll
  for (int it = 0; it < 16; ++it) {
    int ch = it * 256 + tid;                 // 4096 chunks of 4
    int mg = ch & 31, n = ch >> 5;           // n 0..127
    int m = mg * 4;
    float bb = bout[n0g + n];
    float4 t;
    t.x = Cs[(m + 0) * 128 + n] + bb;
    t.y = Cs[(m + 1) * 128 + n] + bb;
    t.z = Cs[(m + 2) * 128 + n] + bb;
    t.w = Cs[(m + 3) * 128 + n] + bb;
    int p = m0 + m;
    int bimg = p >> 12, sp = p & 4095;
    *(float4*)&out[(size_t)bimg * (512 * 4096) + (size_t)(n0g + n) * 4096 + sp] = t;
  }
}

// ------------------------------------------------------------------
extern "C" void kernel_launch(void* const* d_in, const int* in_sizes, int n_in,
                              void* d_out, int out_size, void* d_ws, size_t ws_size,
                              hipStream_t stream) {
  (void)in_sizes; (void)n_in; (void)out_size; (void)ws_size;
  const float* x    = (const float*)d_in[0];
  const float* Wq   = (const float*)d_in[1];
  const float* Wkv  = (const float*)d_in[2];
  const float* Wout = (const float*)d_in[3];
  const float* bout = (const float*)d_in[4];
  const float* relh = (const float*)d_in[5];
  const float* relw = (const float*)d_in[6];

  // workspace: qk (32768x1024 bf16, 64MB) | Vt (8x512x4096 bf16, 32MB) | O (32768x512 bf16, 32MB)
  __bf16* qkb = (__bf16*)d_ws;
  __bf16* Vt  = (__bf16*)((char*)d_ws + (size_t)NPIX * 1024 * 2);
  __bf16* Ob  = (__bf16*)((char*)d_ws + (size_t)NPIX * 1024 * 2 + (size_t)NPIX * 512 * 2);
  float* out = (float*)d_out;

  qkv_proj_kernel<<<dim3(256, 12), 256, 0, stream>>>(x, Wq, Wkv, qkb, Vt);
  halo_attn_kernel<<<dim3(4096), 256, 0, stream>>>(qkb, Vt, relh, relw, Ob);
  out_proj_kernel<<<dim3(256, 4), 256, 0, stream>>>(Ob, Wout, bout, out);
}
